// MambaBlock_3942779978043
// MI455X (gfx1250) — matrix operand order repaired and checked by reading, not compiled, not run on hardware
//
#include <hip/hip_runtime.h>
#include <hip/hip_bf16.h>
#include <cstdint>
#include <cstddef>

// ---------------- problem constants ----------------
#define D_MODEL   256
#define L_SEQ     4096
#define BATCH     8
#define M_TOT     (BATCH * L_SEQ)      // 32768 rows
#define D_INNER   512
#define D_STATE   16
#define NHEADS    8
#define HEADDIM   64
#define CONV_DIM  544                  // D_INNER + 2*D_STATE
#define N_PROJ    1064                 // 2*D_INNER + 2*D_STATE + NHEADS
#define N_PROJ_PAD 1152                // 9 * 128, zero-padded weight rows
#define EPSF      1e-5f

// ---------------- CDNA5 WMMA types ----------------
typedef __attribute__((ext_vector_type(16))) __bf16 v16bf;
typedef __attribute__((ext_vector_type(8)))  __bf16 v8bf;
typedef __attribute__((ext_vector_type(8)))  float  v8f;

static __device__ __forceinline__ float sigmoidf_(float x) {
  return 1.f / (1.f + __expf(-x));
}

// low 32 bits of a flat pointer into LDS == wave-relative LDS address
static __device__ __forceinline__ unsigned lds_addr(const void* p) {
  return (unsigned)(uintptr_t)p;
}

// CDNA5 async global->LDS DMA (ASYNCcnt-tracked; no VGPR data path)
#define ASYNC_LDS_B128(ldsOff, gPtr)                                     \
  asm volatile("global_load_async_to_lds_b128 %0, %1, off" ::"v"(ldsOff), \
               "v"(gPtr)                                                  \
               : "memory")
#define ASYNC_LDS_B64(ldsOff, gPtr)                                     \
  asm volatile("global_load_async_to_lds_b64 %0, %1, off" ::"v"(ldsOff), \
               "v"(gPtr)                                                 \
               : "memory")

// ---------------- fp32 -> bf16 convert ----------------
__global__ __launch_bounds__(256) void k_cvt(const float* __restrict__ s,
                                             __bf16* __restrict__ d, int n) {
  int i = blockIdx.x * 256 + threadIdx.x;
  if (i < n) d[i] = (__bf16)s[i];
}

// convert + zero-pad rows so the GEMM needs no load guards in its hot loop.
__global__ __launch_bounds__(256) void k_cvt_pad(const float* __restrict__ s,
                                                 __bf16* __restrict__ d,
                                                 int n_src, int n_pad) {
  int i = blockIdx.x * 256 + threadIdx.x;
  if (i >= n_pad) return;
  d[i] = (i < n_src) ? (__bf16)s[i] : (__bf16)0.f;
}

// ---------------- LayerNorm (NCHW -> row-major bf16) ----------------
__global__ __launch_bounds__(256) void k_ln(const float* __restrict__ x,
                                            const float* __restrict__ lw,
                                            const float* __restrict__ lb,
                                            __bf16* __restrict__ xn) {
  int b   = blockIdx.y;
  int l0  = blockIdx.x * 32;
  int tid = threadIdx.x;
  __shared__ float sm[D_MODEL * 32];   // [channel][pos], 32 KB

  int cc  = tid >> 5;   // 0..7
  int pos = tid & 31;
  for (int c0 = 0; c0 < D_MODEL; c0 += 8) {
    int c = c0 + cc;
    sm[c * 32 + pos] = x[((size_t)(b * D_MODEL + c)) * L_SEQ + l0 + pos];
  }
  __syncthreads();

  int p2 = tid >> 3;    // position 0..31
  int j  = tid & 7;     // 8 threads reduce one position
  float s = 0.f, ss = 0.f;
  for (int c = j; c < D_MODEL; c += 8) {
    float v = sm[c * 32 + p2];
    s += v; ss += v * v;
  }
  s  += __shfl_xor(s, 1);  ss += __shfl_xor(ss, 1);
  s  += __shfl_xor(s, 2);  ss += __shfl_xor(ss, 2);
  s  += __shfl_xor(s, 4);  ss += __shfl_xor(ss, 4);
  float mu   = s * (1.f / D_MODEL);
  float var  = ss * (1.f / D_MODEL) - mu * mu;
  float rstd = rsqrtf(var + EPSF);

  size_t row = ((size_t)(b * L_SEQ + l0 + p2)) * D_MODEL;
  for (int c = j; c < D_MODEL; c += 8) {
    float v = (sm[c * 32 + p2] - mu) * rstd * lw[c] + lb[c];
    xn[row + c] = (__bf16)v;
  }
}

// ---------------- WMMA fragment helpers ----------------
static __device__ __forceinline__ void load_frags(
    const __bf16* const (&apl)[2], const __bf16* const (&bpl)[4], int k0,
    v16bf (&a)[2], v16bf (&b)[4]) {
#pragma unroll
  for (int i = 0; i < 2; ++i) {
    v8bf lo = *(const v8bf*)(apl[i] + k0);
    v8bf hi = *(const v8bf*)(apl[i] + k0 + 8);
    a[i] = __builtin_shufflevector(lo, hi, 0, 1, 2, 3, 4, 5, 6, 7,
                                   8, 9, 10, 11, 12, 13, 14, 15);
  }
#pragma unroll
  for (int j = 0; j < 4; ++j)
    b[j] = *(const v16bf*)(bpl[j] + k0);
}

static __device__ __forceinline__ void wmma_8(const v16bf (&a)[2],
                                              const v16bf (&b)[4],
                                              v8f (&acc)[2][4]) {
#pragma unroll
  for (int i = 0; i < 2; ++i)
#pragma unroll
    for (int j = 0; j < 4; ++j)
      acc[i][j] = __builtin_amdgcn_wmma_f32_16x16x32_bf16(
          false, a[i], false, b[j], (short)0, acc[i][j], false, false);
}

// ---------------- bf16 WMMA GEMM: C[M x N] = A[M x K] * Bw[Npad x K]^T ----
template <int KD, bool GUARD_STORE>
__global__ __launch_bounds__(256) void k_gemm(const __bf16* __restrict__ A,
                                              const __bf16* __restrict__ Bw,
                                              float* __restrict__ C,
                                              int Nact, int ldc) {
  int tid  = threadIdx.x;
  int wave = tid >> 5, lane = tid & 31;
  int wm = wave & 3, wn = wave >> 2;
  int m0 = blockIdx.y * 128 + wm * 32;
  int n0 = blockIdx.x * 128 + wn * 64;
  int lm = lane & 15;   // row (A) / col (B,D) within 16-tile
  int lh = lane >> 4;   // half-wave select

  const __bf16* apl[2];
#pragma unroll
  for (int i = 0; i < 2; ++i)
    apl[i] = A + ((size_t)(m0 + i * 16 + lm)) * KD + lh * 16;
  const __bf16* bpl[4];
#pragma unroll
  for (int j = 0; j < 4; ++j)
    bpl[j] = Bw + ((size_t)(n0 + j * 16 + lm)) * KD + lh * 16;

  v8f acc[2][4];
#pragma unroll
  for (int i = 0; i < 2; ++i)
#pragma unroll
    for (int j = 0; j < 4; ++j) acc[i][j] = {};

  v16bf a[2], b[4];
  load_frags(apl, bpl, 0, a, b);
#pragma unroll
  for (int k0 = 32; k0 < KD; k0 += 32) {
    v16bf an[2], bn[4];
    load_frags(apl, bpl, k0, an, bn);   // prefetch next step
    wmma_8(a, b, acc);                  // math on current step
#pragma unroll
    for (int i = 0; i < 2; ++i) a[i] = an[i];
#pragma unroll
    for (int j = 0; j < 4; ++j) b[j] = bn[j];
  }
  wmma_8(a, b, acc);

#pragma unroll
  for (int i = 0; i < 2; ++i) {
    int mrow = m0 + i * 16 + lh * 8;
#pragma unroll
    for (int j = 0; j < 4; ++j) {
      int n = n0 + j * 16 + lm;
      if (GUARD_STORE && n >= Nact) continue;
#pragma unroll
      for (int v = 0; v < 8; ++v)
        C[(size_t)(mrow + v) * ldc + n] = acc[i][j][v];
    }
  }
}

// ---------------- GEMM2 + residual + NCHW transpose store -----------------
template <int KD>
__global__ __launch_bounds__(256) void k_gemm_out(const __bf16* __restrict__ A,
                                                  const __bf16* __restrict__ Bw,
                                                  const float* __restrict__ xres,
                                                  float* __restrict__ out) {
  int tid  = threadIdx.x;
  int wave = tid >> 5, lane = tid & 31;
  int wm = wave & 3, wn = wave >> 2;
  int m0 = blockIdx.y * 128 + wm * 32;
  int n0 = blockIdx.x * 128 + wn * 64;
  int lm = lane & 15, lh = lane >> 4;

  const __bf16* apl[2];
#pragma unroll
  for (int i = 0; i < 2; ++i)
    apl[i] = A + ((size_t)(m0 + i * 16 + lm)) * KD + lh * 16;
  const __bf16* bpl[4];
#pragma unroll
  for (int j = 0; j < 4; ++j)
    bpl[j] = Bw + ((size_t)(n0 + j * 16 + lm)) * KD + lh * 16;

  v8f acc[2][4];
#pragma unroll
  for (int i = 0; i < 2; ++i)
#pragma unroll
    for (int j = 0; j < 4; ++j) acc[i][j] = {};

  v16bf a[2], b[4];
  load_frags(apl, bpl, 0, a, b);
#pragma unroll
  for (int k0 = 32; k0 < KD; k0 += 32) {
    v16bf an[2], bn[4];
    load_frags(apl, bpl, k0, an, bn);
    wmma_8(a, b, acc);
#pragma unroll
    for (int i = 0; i < 2; ++i) a[i] = an[i];
#pragma unroll
    for (int j = 0; j < 4; ++j) b[j] = bn[j];
  }
  wmma_8(a, b, acc);

#pragma unroll
  for (int i = 0; i < 2; ++i) {
    int mrow = m0 + i * 16 + lh * 8;
    int bb = mrow >> 12;           // / 4096
    int l  = mrow & (L_SEQ - 1);
#pragma unroll
    for (int j = 0; j < 4; ++j) {
      int n = n0 + j * 16 + lm;    // output channel
      size_t base = ((size_t)(bb * D_MODEL + n)) * L_SEQ + l;
#pragma unroll
      for (int v = 0; v < 8; ++v)
        out[base + v] = acc[i][j][v] + xres[base + v];
    }
  }
}

// ---------------- depthwise causal conv (k=4) + SiLU ----------------
__global__ __launch_bounds__(256) void k_conv(const float* __restrict__ zx,
                                              const float* __restrict__ cw,
                                              const float* __restrict__ cb,
                                              float* __restrict__ xc) {
  int idx = blockIdx.x * 256 + threadIdx.x;
  if (idx >= M_TOT * CONV_DIM) return;
  int c = idx % CONV_DIM;
  int m = idx / CONV_DIM;
  int l = m & (L_SEQ - 1);
  float acc = cb[c];
#pragma unroll
  for (int k = 0; k < 4; ++k) {
    int lk = l - 3 + k;
    if (lk >= 0)
      acc += zx[(size_t)(m - 3 + k) * N_PROJ + D_INNER + c] * cw[c * 4 + k];
  }
  xc[(size_t)m * CONV_DIM + c] = acc * sigmoidf_(acc);   // SiLU
}

// ---------------- dt: softplus + dA, interleaved (dt,dA) per (m,h) -------
// datc[m*16 + h*2 + 0] = softplus(dt), [.. + 1] = exp(dt*A): one b64 async
// load in the scan fetches both.
__global__ __launch_bounds__(256) void k_dt(const float* __restrict__ zx,
                                            const float* __restrict__ dtb,
                                            const float* __restrict__ alog,
                                            float* __restrict__ datc) {
  int idx = blockIdx.x * 256 + threadIdx.x;
  if (idx >= M_TOT * NHEADS) return;
  int h = idx & 7;
  int m = idx >> 3;
  float v  = zx[(size_t)m * N_PROJ + (N_PROJ - NHEADS) + h] + dtb[h];
  float sp = (v > 20.f) ? v : log1pf(__expf(v));
  datc[(size_t)m * 16 + h * 2 + 0] = sp;
  datc[(size_t)m * 16 + h * 2 + 1] = __expf(-sp * __expf(alog[h]));
}

// ---------------- sequential SSM scan ----------------
// Block = one (batch, head). 1024 threads: tid = p*16 + n holds h[p][n] in a
// register. 64-step chunks double-buffered in LDS and filled with
// global_load_async_to_lds (ASYNCcnt): chunk k+1's DMA overlaps chunk k's
// FMA chain. Waves 0-1 issue 3 asyncs/chunk, waves 2-31 issue 2, so the
// pipelined s_wait_asynccnt immediate is wave-uniform.
__global__ __launch_bounds__(1024) void k_scan(const float* __restrict__ xc,
                                               const float* __restrict__ datc,
                                               const float* __restrict__ Dp,
                                               float* __restrict__ y) {
  int b = blockIdx.x >> 3;
  int h = blockIdx.x & 7;
  int tid = threadIdx.x;
  int p = tid >> 4, n = tid & 15;

  __shared__ float sX[2][64 * 64];    // [buf][step*64 + p]      32 KB
  __shared__ float sBC[2][64 * 32];   // [buf][step*32 + (B:0-15 C:16-31)] 16 KB
  __shared__ float sDT[2][64 * 2];    // [buf][step*2 + (dt,dA)]  1 KB

  float hs = 0.f;
  float Dh = Dp[h];

  // issue one chunk's async loads (all threads; wave-uniform counts)
  auto issue = [&](int buf, int t0) {
    int s  = tid >> 4;          // 0..63 step
    int q4 = (tid & 15) * 4;    // 4 floats per lane
    const float* gx =
        &xc[(size_t)(b * L_SEQ + t0 + s) * CONV_DIM + h * HEADDIM + q4];
    ASYNC_LDS_B128(lds_addr(&sX[buf][s * 64 + q4]), gx);
    int q2 = (tid & 15) * 2;    // 2 floats per lane -> 32 (B||C) per step
    const float* gbc =
        &xc[(size_t)(b * L_SEQ + t0 + s) * CONV_DIM + D_INNER + q2];
    ASYNC_LDS_B64(lds_addr(&sBC[buf][s * 32 + q2]), gbc);
    if (tid < 64) {             // waves 0-1: (dt,dA) pair per step
      const float* gdt = &datc[(size_t)(b * L_SEQ + t0 + tid) * 16 + h * 2];
      ASYNC_LDS_B64(lds_addr(&sDT[buf][tid * 2]), gdt);
    }
  };

  issue(0, 0);
  const int NCHUNK = L_SEQ / 64;
  for (int chunk = 0; chunk < NCHUNK; ++chunk) {
    int cur = chunk & 1;
    if (chunk + 1 < NCHUNK) {
      issue(cur ^ 1, (chunk + 1) * 64);
      // wait until only the just-issued chunk's loads are outstanding
      if (tid < 64)
        asm volatile("s_wait_asynccnt 0x3" ::: "memory");
      else
        asm volatile("s_wait_asynccnt 0x2" ::: "memory");
    } else {
      asm volatile("s_wait_asynccnt 0x0" ::: "memory");
    }
    __syncthreads();   // LDS chunk visible to all waves

    int t0 = chunk * 64;
    for (int s = 0; s < 64; ++s) {
      float tt = sDT[cur][s * 2 + 0];
      float aa = sDT[cur][s * 2 + 1];
      float xv = sX[cur][s * 64 + p];
      hs = aa * hs + (tt * xv) * sBC[cur][s * 32 + n];
      float yp = hs * sBC[cur][s * 32 + 16 + n];
      yp += __shfl_xor(yp, 1);
      yp += __shfl_xor(yp, 2);
      yp += __shfl_xor(yp, 4);
      yp += __shfl_xor(yp, 8);
      if (n == 0)
        y[(size_t)(b * L_SEQ + t0 + s) * D_INNER + h * HEADDIM + p] =
            yp + Dh * xv;
    }
    __syncthreads();   // done with 'cur' before it is overwritten
  }
}

// ---------------- SiLU gate + RMSNorm -> bf16 ----------------
__global__ __launch_bounds__(256) void k_gate(const float* __restrict__ zx,
                                              const float* __restrict__ y,
                                              const float* __restrict__ rw,
                                              __bf16* __restrict__ yb) {
  int m = blockIdx.x;
  int tid = threadIdx.x;
  size_t zrow = (size_t)m * N_PROJ;
  size_t yrow = (size_t)m * D_INNER;

  float g[2];
  float gg = 0.f;
#pragma unroll
  for (int r = 0; r < 2; ++r) {
    int c = tid + r * 256;
    float z = zx[zrow + c];
    float v = y[yrow + c] * (z * sigmoidf_(z));
    g[r] = v;
    gg += v * v;
  }
  gg += __shfl_xor(gg, 1);
  gg += __shfl_xor(gg, 2);
  gg += __shfl_xor(gg, 4);
  gg += __shfl_xor(gg, 8);
  gg += __shfl_xor(gg, 16);

  __shared__ float red[8];
  if ((tid & 31) == 0) red[tid >> 5] = gg;
  __syncthreads();
  float tot = 0.f;
#pragma unroll
  for (int w = 0; w < 8; ++w) tot += red[w];
  float rr = rsqrtf(tot * (1.f / D_INNER) + EPSF);

#pragma unroll
  for (int r = 0; r < 2; ++r) {
    int c = tid + r * 256;
    yb[yrow + c] = (__bf16)(g[r] * rr * rw[c]);
  }
}

// ---------------- host launcher ----------------
extern "C" void kernel_launch(void* const* d_in, const int* in_sizes, int n_in,
                              void* d_out, int out_size, void* d_ws,
                              size_t ws_size, hipStream_t stream) {
  (void)in_sizes; (void)n_in; (void)out_size; (void)ws_size;

  const float* x    = (const float*)d_in[0];
  const float* lw   = (const float*)d_in[1];
  const float* lb   = (const float*)d_in[2];
  const float* w1   = (const float*)d_in[3];
  const float* cw   = (const float*)d_in[4];
  const float* cb   = (const float*)d_in[5];
  const float* dtb  = (const float*)d_in[6];
  const float* alog = (const float*)d_in[7];
  const float* Dp   = (const float*)d_in[8];
  const float* rw   = (const float*)d_in[9];
  const float* w2   = (const float*)d_in[10];
  float* out = (float*)d_out;

  char* ws = (char*)d_ws;
  size_t off = 0;
  auto alloc = [&](size_t bytes) {
    size_t o = off;
    off += (bytes + 255) & ~(size_t)255;
    return o;
  };
  __bf16* w1b  = (__bf16*)(ws + alloc((size_t)N_PROJ_PAD * D_MODEL * 2));
  __bf16* w2b  = (__bf16*)(ws + alloc((size_t)D_MODEL * D_INNER * 2));
  __bf16* xnb  = (__bf16*)(ws + alloc((size_t)M_TOT * D_MODEL * 2));
  float*  zx   = (float*) (ws + alloc((size_t)M_TOT * N_PROJ * 4));
  float*  xc   = (float*) (ws + alloc((size_t)M_TOT * CONV_DIM * 4));
  float*  datc = (float*) (ws + alloc((size_t)M_TOT * 16 * 4));
  float*  yss  = (float*) (ws + alloc((size_t)M_TOT * D_INNER * 4));
  __bf16* ybf  = (__bf16*)(ws + alloc((size_t)M_TOT * D_INNER * 2));

  // weights to bf16; W1 zero-padded to 1152 rows -> guard-free GEMM loop
  k_cvt_pad<<<(N_PROJ_PAD * D_MODEL + 255) / 256, 256, 0, stream>>>(
      w1, w1b, N_PROJ * D_MODEL, N_PROJ_PAD * D_MODEL);
  k_cvt<<<(D_MODEL * D_INNER + 255) / 256, 256, 0, stream>>>(w2, w2b,
                                                             D_MODEL * D_INNER);
  // LayerNorm + layout change + bf16 cast
  k_ln<<<dim3(L_SEQ / 32, BATCH), 256, 0, stream>>>(x, lw, lb, xnb);
  // in_proj GEMM (N padded to 9*128; stores guarded at n >= 1064)
  k_gemm<D_MODEL, true><<<dim3(N_PROJ_PAD / 128, M_TOT / 128), 256, 0,
                          stream>>>(xnb, w1b, zx, N_PROJ, N_PROJ);
  // depthwise conv + SiLU
  k_conv<<<(M_TOT * CONV_DIM + 255) / 256, 256, 0, stream>>>(zx, cw, cb, xc);
  // dt softplus + dA (interleaved pairs for the scan's b64 async loads)
  k_dt<<<(M_TOT * NHEADS + 255) / 256, 256, 0, stream>>>(zx, dtb, alog, datc);
  // sequential SSM scan with double-buffered async-to-LDS chunks
  k_scan<<<BATCH * NHEADS, 1024, 0, stream>>>(xc, datc, Dp, yss);
  // gate + RMSNorm -> bf16
  k_gate<<<M_TOT, 256, 0, stream>>>(zx, yss, rw, ybf);
  // out_proj GEMM + residual + NCHW store
  k_gemm_out<D_INNER><<<dim3(D_MODEL / 128, M_TOT / 128), 256, 0, stream>>>(
      ybf, w2b, x, out);
}
